// MultiHeadAttention_20066087207431
// MI455X (gfx1250) — compile-verified, matrix operand order checked by reading, not run on hardware
//
#include <hip/hip_runtime.h>
#include <hip/hip_bf16.h>

typedef __attribute__((ext_vector_type(16))) _Float16 v16h;
typedef __attribute__((ext_vector_type(8)))  _Float16 v8h;
typedef __attribute__((ext_vector_type(4)))  _Float16 v4h;
typedef __attribute__((ext_vector_type(8)))  float    v8f;
typedef __attribute__((ext_vector_type(4)))  float    v4f;

#define H_   16
#define DH_  64
#define NV_  1024
#define B_   2
#define S_   2048
#define BH_  (B_ * H_)
#define BS_  (B_ * S_)

static __device__ __forceinline__ void cvt_store4(_Float16* dst, v4f f)
{
    v4h h;
    h[0] = (_Float16)f[0]; h[1] = (_Float16)f[1];
    h[2] = (_Float16)f[2]; h[3] = (_Float16)f[3];
    *(v4h*)dst = h;                                  // ds_store_b64
}

// ---------------------------------------------------------------------------
// Unified GEMM:  C[M x 1024] = A[M x 1024] * W[1024 x 1024]^T + bias
//   A, W fp32 row-major; converted to f16 while staging into LDS.
//   Double-buffered LDS; next tile's global_load_b128s are issued before the
//   WMMA block so they are in flight during compute (no per-load waits).
//   mode 0: write fp32 row-major to Cf               (final output projection)
//   mode 1: write f16 head-split Ch[((b*16+h)*S+s)*64+d]   (Q/K/V projections)
// Block: 256 thr (8 waves). C tile 128x64. Wave grid 4x2, wave tile 32x32
// (2x2 v_wmma_f32_16x16x32_f16 tiles). K-step 32.
// ---------------------------------------------------------------------------
__global__ __launch_bounds__(256) void gemm_wmma_f16(
    const float* __restrict__ A, const float* __restrict__ W,
    const float* __restrict__ bias, float* __restrict__ Cf,
    _Float16* __restrict__ Ch, int M, int mode)
{
    __shared__ _Float16 As[2][128][32];
    __shared__ _Float16 Bs[2][64][32];

    const int tid   = threadIdx.x;
    const int lane  = tid & 31;
    const int wave  = tid >> 5;
    const int m0    = blockIdx.y * 128;
    const int n0    = blockIdx.x * 64;
    const int mBase = (wave >> 1) * 32;   // 0,32,64,96
    const int nBase = (wave & 1) * 32;    // 0,32
    const int lr    = lane & 15;          // M (A frag) / N (B frag)
    const int hi    = lane >> 4;
    const int kb    = hi * 8;             // lo lanes: K 0..7 & 16..23 ; hi: +8

    // per-thread staging coordinates (constant across K-steps)
    const int amRow[4] = { (tid + 0)   >> 3, (tid + 256) >> 3,
                           (tid + 512) >> 3, (tid + 768) >> 3 };
    const int ak4     = (tid & 7) * 4;
    const int wnRow[2] = { (tid + 0) >> 3, (tid + 256) >> 3 };

    v8f acc[2][2] = {};
    v4f fa[4], fw[2];

    // ---- prologue: load + stage tile kt=0 into buffer 0 ----
    #pragma unroll
    for (int it = 0; it < 4; ++it)
        fa[it] = *(const v4f*)(A + (size_t)(m0 + amRow[it]) * NV_ + ak4);
    #pragma unroll
    for (int it = 0; it < 2; ++it)
        fw[it] = *(const v4f*)(W + (size_t)(n0 + wnRow[it]) * NV_ + ak4);
    #pragma unroll
    for (int it = 0; it < 4; ++it) cvt_store4(&As[0][amRow[it]][ak4], fa[it]);
    #pragma unroll
    for (int it = 0; it < 2; ++it) cvt_store4(&Bs[0][wnRow[it]][ak4], fw[it]);
    __syncthreads();

    int cur = 0;
    for (int kt = 0;; kt += 32) {
        const bool hasNext = (kt + 32 < NV_);
        if (hasNext) {
            // issue next tile's loads now; consumed only after the WMMAs
            #pragma unroll
            for (int it = 0; it < 4; ++it)
                fa[it] = *(const v4f*)(A + (size_t)(m0 + amRow[it]) * NV_ + kt + 32 + ak4);
            #pragma unroll
            for (int it = 0; it < 2; ++it)
                fw[it] = *(const v4f*)(W + (size_t)(n0 + wnRow[it]) * NV_ + kt + 32 + ak4);
            if (kt + 64 < NV_) {   // HBM->L2 prefetch two tiles ahead
                if (tid < 128)      __builtin_prefetch(&A[(size_t)(m0 + tid) * NV_ + kt + 64], 0, 1);
                else if (tid < 192) __builtin_prefetch(&W[(size_t)(n0 + tid - 128) * NV_ + kt + 64], 0, 1);
            }
        }

        // ---- fragments: two contiguous 8-halve runs per lane (b128 x2) ----
        v16h a[2], b[2];
        #pragma unroll
        for (int t = 0; t < 2; t++) {
            const _Float16* rp = &As[cur][mBase + 16 * t + lr][0];
            #pragma unroll
            for (int i = 0; i < 8; i++) { a[t][i] = rp[kb + i]; a[t][8 + i] = rp[kb + 16 + i]; }
        }
        #pragma unroll
        for (int t = 0; t < 2; t++) {
            const _Float16* rp = &Bs[cur][nBase + 16 * t + lr][0];
            #pragma unroll
            for (int i = 0; i < 8; i++) { b[t][i] = rp[kb + i]; b[t][8 + i] = rp[kb + 16 + i]; }
        }
        #pragma unroll
        for (int mt = 0; mt < 2; mt++)
            #pragma unroll
            for (int nt = 0; nt < 2; nt++)
                acc[mt][nt] = __builtin_amdgcn_wmma_f32_16x16x32_f16(
                    false, a[mt], false, b[nt], (short)0, acc[mt][nt], false, false);

        if (!hasNext) break;

        // stage next tile into the other buffer; its previous readers were
        // already separated by the prior barrier -> one barrier per step.
        const int nb = cur ^ 1;
        #pragma unroll
        for (int it = 0; it < 4; ++it) cvt_store4(&As[nb][amRow[it]][ak4], fa[it]);
        #pragma unroll
        for (int it = 0; it < 2; ++it) cvt_store4(&Bs[nb][wnRow[it]][ak4], fw[it]);
        __syncthreads();
        cur = nb;
    }

    // ---- epilogue: C layout lane(0-15): M=r ; lane(16-31): M=r+8 ; N=lr ----
    #pragma unroll
    for (int nt = 0; nt < 2; nt++) {
        const int ng = n0 + nBase + nt * 16 + lr;
        const float bv = bias[ng];
        #pragma unroll
        for (int mt = 0; mt < 2; mt++) {
            #pragma unroll
            for (int r = 0; r < 8; r++) {
                const int mg = m0 + mBase + mt * 16 + r + 8 * hi;
                const float v = acc[mt][nt][r] + bv;
                if (mode == 0) {
                    Cf[(size_t)mg * NV_ + ng] = v;
                } else {
                    const int h = ng >> 6, d = ng & 63;
                    const int bb = mg >> 11, s = mg & (S_ - 1);
                    Ch[(((size_t)(bb * H_ + h)) * S_ + s) * DH_ + d] = (_Float16)v;
                }
            }
        }
    }
}

// ---------------------------------------------------------------------------
// Per (b,h):  M = (K1^T @ V1) / 8   -> 64x64 fp32
// Block: 128 thr (4 waves); wave w owns C rows 16w..16w+15 (4 N-tiles).
// K/V tiles staged TRANSPOSED into LDS; loads issued as a batch (b128) before
// the scatter so they overlap.
// ---------------------------------------------------------------------------
__global__ __launch_bounds__(128) void ktv_wmma(
    const _Float16* __restrict__ Kh, const _Float16* __restrict__ Vh,
    float* __restrict__ Mb)
{
    __shared__ _Float16 Kt[64][32];   // [d][s_local]
    __shared__ _Float16 Vt[64][32];

    const int bh   = blockIdx.x;
    const int tid  = threadIdx.x;
    const int lane = tid & 31;
    const int wave = tid >> 5;
    const int lr   = lane & 15;
    const int hi   = lane >> 4;
    const int kb   = hi * 8;

    const _Float16* kp = Kh + (size_t)bh * S_ * DH_;
    const _Float16* vp = Vh + (size_t)bh * S_ * DH_;

    // staging coords: 32 rows x 64 cols = 256 8-halve chunks, 2 per thread
    const int rRow[2] = { (tid + 0) >> 3, (tid + 128) >> 3 };
    const int c8      = (tid & 7) * 8;

    v8f acc[4] = {};

    for (int st = 0; st < S_; st += 32) {
        v8h rk[2], rv[2];
        #pragma unroll
        for (int it = 0; it < 2; ++it) {          // 4 global_load_b128 in flight
            rk[it] = *(const v8h*)(kp + (size_t)(st + rRow[it]) * DH_ + c8);
            rv[it] = *(const v8h*)(vp + (size_t)(st + rRow[it]) * DH_ + c8);
        }
        #pragma unroll
        for (int it = 0; it < 2; ++it)
            #pragma unroll
            for (int j = 0; j < 8; j++) {
                Kt[c8 + j][rRow[it]] = rk[it][j];
                Vt[c8 + j][rRow[it]] = rv[it][j];
            }
        __syncthreads();

        const _Float16* ap = &Kt[wave * 16 + lr][0]; // A = K^T, row = d-of-K
        v16h a;
        #pragma unroll
        for (int i = 0; i < 8; i++) { a[i] = ap[kb + i]; a[8 + i] = ap[kb + 16 + i]; }

        #pragma unroll
        for (int nt = 0; nt < 4; nt++) {
            const _Float16* bp = &Vt[nt * 16 + lr][0];  // B^T = V^T, row = d-of-V
            v16h b;
            #pragma unroll
            for (int i = 0; i < 8; i++) { b[i] = bp[kb + i]; b[8 + i] = bp[kb + 16 + i]; }
            acc[nt] = __builtin_amdgcn_wmma_f32_16x16x32_f16(
                false, a, false, b, (short)0, acc[nt], false, false);
        }
        __syncthreads();
    }

    #pragma unroll
    for (int nt = 0; nt < 4; nt++)
        #pragma unroll
        for (int r = 0; r < 8; r++)
            Mb[(size_t)bh * 4096 + (wave * 16 + r + 8 * hi) * 64 + nt * 16 + lr]
                = acc[nt][r] * 0.125f;
}

// ---------------------------------------------------------------------------
// X[b][s][h*64+d] = softmax_d( Q1[bh][s][:] @ M[bh] )     (fp32, row over 64)
// One thread per (bh, s) row; M broadcast from LDS.
// ---------------------------------------------------------------------------
__global__ __launch_bounds__(256) void attn_softmax(
    const _Float16* __restrict__ Qh, const float* __restrict__ Mb,
    float* __restrict__ X)
{
    __shared__ float Ms[64 * 64];
    const int bh  = blockIdx.x;
    const int tid = threadIdx.x;

    #pragma unroll
    for (int it = 0; it < 16; ++it) {
        const int idx = tid + it * 256;
        Ms[idx] = Mb[(size_t)bh * 4096 + idx];
    }
    __syncthreads();

    const int s = blockIdx.y * 256 + tid;
    const v8h* qp8 = (const v8h*)(Qh + ((size_t)bh * S_ + s) * DH_);  // 128B aligned

    float q[64];
    #pragma unroll
    for (int c = 0; c < 8; c++) {
        const v8h qv = qp8[c];                    // global_load_b128
        #pragma unroll
        for (int j = 0; j < 8; j++) q[c * 8 + j] = (float)qv[j];
    }

    float lg[64];
    #pragma unroll
    for (int d = 0; d < 64; d++) lg[d] = 0.f;
    for (int e = 0; e < 64; e++) {
        const float qe = q[e];
        #pragma unroll
        for (int d = 0; d < 64; d++) lg[d] = fmaf(qe, Ms[e * 64 + d], lg[d]);
    }

    float mx = lg[0];
    #pragma unroll
    for (int d = 1; d < 64; d++) mx = fmaxf(mx, lg[d]);
    float sum = 0.f;
    #pragma unroll
    for (int d = 0; d < 64; d++) { lg[d] = __expf(lg[d] - mx); sum += lg[d]; }
    const float inv = 1.0f / sum;

    const int bb = bh >> 4, h = bh & 15;
    float* xp = X + ((size_t)bb * S_ + s) * NV_ + h * DH_;
    #pragma unroll
    for (int d = 0; d < 64; d++) xp[d] = lg[d] * inv;
}

// ---------------------------------------------------------------------------
extern "C" void kernel_launch(void* const* d_in, const int* in_sizes, int n_in,
                              void* d_out, int out_size, void* d_ws, size_t ws_size,
                              hipStream_t stream)
{
    (void)in_sizes; (void)n_in; (void)out_size; (void)ws_size;
    const float* Sq    = (const float*)d_in[0];
    const float* Sk    = (const float*)d_in[1];
    const float* Sv    = (const float*)d_in[2];
    const float* Wq_w  = (const float*)d_in[3];
    const float* Wq_b  = (const float*)d_in[4];
    const float* out_w = (const float*)d_in[5];
    const float* out_b = (const float*)d_in[6];
    float* Out = (float*)d_out;

    // workspace: Qh/Kh/Vh f16 (3 x 8MB) | M fp32 (0.5MB) | X fp32 (16MB)
    _Float16* Qh = (_Float16*)d_ws;
    _Float16* Kh = Qh + (size_t)BH_ * S_ * DH_;
    _Float16* Vh = Kh + (size_t)BH_ * S_ * DH_;
    float*    Mb = (float*)(Vh + (size_t)BH_ * S_ * DH_);
    float*    X  = Mb + (size_t)BH_ * DH_ * DH_;

    dim3 gGemm(NV_ / 64, BS_ / 128);   // 16 x 32

    gemm_wmma_f16<<<gGemm, 256, 0, stream>>>(Sq, Wq_w, Wq_b, nullptr, Qh, BS_, 1);
    gemm_wmma_f16<<<gGemm, 256, 0, stream>>>(Sk, Wq_w, Wq_b, nullptr, Kh, BS_, 1);
    gemm_wmma_f16<<<gGemm, 256, 0, stream>>>(Sv, Wq_w, Wq_b, nullptr, Vh, BS_, 1);
    ktv_wmma<<<BH_, 128, 0, stream>>>(Kh, Vh, Mb);
    attn_softmax<<<dim3(BH_, S_ / 256), 256, 0, stream>>>(Qh, Mb, X);
    gemm_wmma_f16<<<gGemm, 256, 0, stream>>>(X, out_w, out_b, Out, nullptr, BS_, 0);
}
